// TaskAwareMoE_25099788878583
// MI455X (gfx1250) — matrix operand order repaired and checked by reading, not compiled, not run on hardware
//
#include <hip/hip_runtime.h>
#include <hip/hip_bf16.h>

// Problem constants (match reference)
constexpr int Bc = 8;
constexpr int Nc = 2048;
constexpr int Dc = 512;
constexpr int Ec = 8;
constexpr int NTOK = Bc * Nc;          // 16384

// Tiling for GEMM kernels: 128 threads = 4 wave32, each wave owns 32 rows
constexpr int BM = 128;                 // rows (tokens) per workgroup
constexpr int BN = 64;                  // output cols per workgroup
constexpr int BK = 32;                  // K per step (WMMA bf16 K)
constexpr int KSTEPS = Dc / BK;         // 16

typedef __attribute__((ext_vector_type(16))) __bf16 v16bf;
typedef __attribute__((ext_vector_type(8)))  float  v8f;
typedef __attribute__((ext_vector_type(4)))  unsigned int u32x4;

union Frag { v16bf v; u32x4 q[2]; };

__device__ __forceinline__ float gelu_exact(float v) {
    return 0.5f * v * (1.0f + erff(v * 0.70710678118654752440f));
}

// gfx1250 async global->LDS copy (GVS mode: SGPR base + 32-bit VGPR offset),
// tracked by ASYNCcnt, no VGPR data transit.
__device__ __forceinline__ void async_copy_b128_s(void* lds, const void* sbase, unsigned voff) {
    unsigned l = (unsigned)(uintptr_t)lds;                 // low 32 bits = LDS offset
    asm volatile("global_load_async_to_lds_b128 %0, %1, %2"
                 :: "v"(l), "v"(voff), "s"((unsigned long long)(uintptr_t)sbase)
                 : "memory");
}
__device__ __forceinline__ void wait_async0() {
    asm volatile("s_wait_asynccnt 0x0" ::: "memory");
}

// ---------------------------------------------------------------- convert x
__global__ __launch_bounds__(256) void k_convert_x(const float* __restrict__ x,
                                                   __hip_bfloat16* __restrict__ xb) {
    int i = (blockIdx.x * 256 + threadIdx.x) * 4;   // total NTOK*Dc elems
    #pragma unroll
    for (int t = 0; t < 4; ++t) xb[i + t] = __float2bfloat16(x[i + t]);
}

// ------------------------------------------- convert + transpose weights
// wt[e][n][k] = (e<8 ? W_experts[e][k][n] : W_univ[k][n]), bf16
__global__ __launch_bounds__(256) void k_convert_w(const float* __restrict__ We,
                                                   const float* __restrict__ Wu,
                                                   __hip_bfloat16* __restrict__ wt) {
    int idx = blockIdx.x * 256 + threadIdx.x;       // total 9*512*512
    int e   = idx >> 18;
    int rem = idx & 262143;
    int n   = rem >> 9;
    int k   = rem & 511;
    float v = (e < 8) ? We[(size_t)e * 262144 + (size_t)k * 512 + n]
                      : Wu[(size_t)k * 512 + n];
    wt[idx] = __float2bfloat16(v);
}

// ---------------------------------------------------------------- gating GEMM
__global__ __launch_bounds__(256) void k_gate(const float* __restrict__ x,
                                              const int*   __restrict__ task_ids,
                                              const float* __restrict__ task_embed,
                                              const float* __restrict__ Wg,
                                              const float* __restrict__ bg,
                                              float* __restrict__ logits) {
    int g   = blockIdx.x * 256 + threadIdx.x;       // NTOK*8 threads
    int tok = g >> 3;
    int e   = g & 7;
    int b   = tok >> 11;
    const float* xr = x + (size_t)tok * Dc;
    const float* tr = task_embed + (size_t)task_ids[b] * Dc;
    float acc = bg[e];
    for (int d = 0; d < Dc; ++d) {
        acc += xr[d] * Wg[d * Ec + e];
        acc += tr[d] * Wg[(Dc + d) * Ec + e];
    }
    logits[(size_t)tok * Ec + e] = acc;
}

// ---------------------------------------------------------------- zero counts
__global__ void k_zero(unsigned* counts) {
    if (threadIdx.x < Ec) counts[threadIdx.x] = 0u;
}

// --------------------------------------------- top-2 gates + expert routing
__global__ __launch_bounds__(256) void k_top2(const float* __restrict__ logits,
                                              float* __restrict__ gates,
                                              float* __restrict__ omega,
                                              unsigned* __restrict__ counts,
                                              int* __restrict__ tokList) {
    int tok = blockIdx.x * 256 + threadIdx.x;       // NTOK threads
    float v[8];
    #pragma unroll
    for (int e = 0; e < 8; ++e) v[e] = logits[(size_t)tok * Ec + e];
    int i0 = 0;
    #pragma unroll
    for (int e = 1; e < 8; ++e) if (v[e] > v[i0]) i0 = e;
    int i1 = (i0 == 0) ? 1 : 0;
    #pragma unroll
    for (int e = 0; e < 8; ++e) if (e != i0 && v[e] > v[i1]) i1 = e;
    float ex = expf(v[i1] - v[i0]);                 // v[i0] >= v[i1]
    float g0 = 1.0f / (1.0f + ex);
    float g1 = ex * g0;
    #pragma unroll
    for (int e = 0; e < 8; ++e) gates[(size_t)tok * Ec + e] = 0.0f;
    gates[(size_t)tok * Ec + i0] = g0;
    gates[(size_t)tok * Ec + i1] = g1;
    omega[tok] = 1.0f - g0;
    // routing lists (slot order is nondeterministic but does not affect output)
    unsigned s0 = atomicAdd(&counts[i0], 1u);
    tokList[(size_t)i0 * NTOK + s0] = tok;
    unsigned s1 = atomicAdd(&counts[i1], 1u);
    tokList[(size_t)i1 * NTOK + s1] = tok;
}

// ------------------------------------------------ WMMA GEMM (dense or routed)
// GATHER=false: rows = all tokens, out = omega * gelu(x@Wu + b)      (write)
// GATHER=true : rows = tokList[expert], out += gate * gelu(x@We + b) (guarded RMW)
// grid = (NTOK/BM, Dc/BN), block = 128 (4 wave32), wave w owns rows [32w,32w+32)
template <bool GATHER>
__global__ __launch_bounds__(128) void k_gemm(
        const unsigned char* __restrict__ xb,       // bf16 [NTOK][512]
        const unsigned char* __restrict__ wtm,      // bf16 [512(n)][512(k)] of this matrix
        const float* __restrict__ gates,            // [NTOK][8]
        const float* __restrict__ omega,            // [NTOK]
        const float* __restrict__ bias,             // [512] of this matrix
        const unsigned* __restrict__ counts,        // [8]
        const int* __restrict__ tokList,            // [8][NTOK]
        int expert,
        float* __restrict__ out) {                  // [NTOK][512]

    __shared__ alignas(16) unsigned char sA[2][BM * BK * 2];  // 2 x 8192 B
    __shared__ alignas(16) unsigned char sB[2][BN * BK * 2];  // 2 x 4096 B

    const int tid  = threadIdx.x;
    const int wave = tid >> 5;       // 0..3
    const int lane = tid & 31;
    const int hi   = lane >> 4;
    const int ln   = lane & 15;

    const int tBase = blockIdx.x * BM;
    const int n0    = blockIdx.y * BN;

    int count = NTOK;
    const int* list = nullptr;
    if (GATHER) {
        count = (int)counts[expert];
        if (tBase >= count) return;                 // early-exit spare tiles
        list = tokList + (size_t)expert * NTOK;
    }
    auto rowToken = [&](int idx) -> int {           // clamped gather (pad rows)
        int ci = (idx < count) ? idx : (count - 1);
        return GATHER ? list[ci] : ci;
    };

    // per-thread staging assignments (128 threads: 4 A-chunks + 2 B-chunks each)
    const int aR = tid >> 2, aC = tid & 3;          // aR in 0..31
    unsigned aOff[4];
    #pragma unroll
    for (int rep = 0; rep < 4; ++rep)
        aOff[rep] = (unsigned)rowToken(tBase + aR + 32 * rep) * (Dc * 2) + aC * 16;
    const int bR = aR, bC = aC;
    unsigned bOff[2];
    #pragma unroll
    for (int rep = 0; rep < 2; ++rep)
        bOff[rep] = (unsigned)(n0 + bR + 32 * rep) * (Dc * 2) + bC * 16;

    auto issue = [&](int kt, int buf) {
        const unsigned ko = (unsigned)kt * (BK * 2);
        #pragma unroll
        for (int rep = 0; rep < 4; ++rep)
            async_copy_b128_s(sA[buf] + (aR + 32 * rep) * 64 + aC * 16, xb, aOff[rep] + ko);
        #pragma unroll
        for (int rep = 0; rep < 2; ++rep)
            async_copy_b128_s(sB[buf] + (bR + 32 * rep) * 64 + bC * 16, wtm, bOff[rep] + ko);
    };

    v8f acc[2][4];
    #pragma unroll
    for (int mf = 0; mf < 2; ++mf)
        #pragma unroll
        for (int f = 0; f < 4; ++f)
            #pragma unroll
            for (int r = 0; r < 8; ++r) acc[mf][f][r] = 0.0f;

    issue(0, 0);
    for (int kt = 0; kt < KSTEPS; ++kt) {
        const int cur = kt & 1;
        wait_async0();                               // my tile-kt copies arrived
        __syncthreads();                             // everyone's arrived; prev compute done
        if (kt + 1 < KSTEPS) issue(kt + 1, cur ^ 1); // prefetch next tile (overlaps WMMA)

        // A frags: row M = ln; lanes<16 hold K{0..7,16..23}, lanes>=16 hold K{8..15,24..31}
        Frag a[2];
        #pragma unroll
        for (int mf = 0; mf < 2; ++mf) {
            const unsigned char* arow = sA[cur] + (wave * 32 + mf * 16 + ln) * 64;
            a[mf].q[0] = *(const u32x4*)(arow + hi * 16);
            a[mf].q[1] = *(const u32x4*)(arow + hi * 16 + 32);
        }
        #pragma unroll
        for (int f = 0; f < 4; ++f) {
            // B frag: col N = f*16+ln; lanes<16 hold K{0..7,16..23}, lanes>=16 hold K{8..15,24..31}
            Frag bfr;
            const unsigned char* brow = sB[cur] + (f * 16 + ln) * 64;
            bfr.q[0] = *(const u32x4*)(brow + hi * 16);
            bfr.q[1] = *(const u32x4*)(brow + hi * 16 + 32);
            #pragma unroll
            for (int mf = 0; mf < 2; ++mf)
                acc[mf][f] = __builtin_amdgcn_wmma_f32_16x16x32_bf16(
                    false, a[mf].v, false, bfr.v, (short)0, acc[mf][f], false, false);
        }
    }

    // epilogue: bias + exact GELU + gate/omega weighting
    #pragma unroll
    for (int mf = 0; mf < 2; ++mf) {
        const int rBase = tBase + wave * 32 + mf * 16 + hi * 8;  // C/D: M = r + 8*hi
        float wrow[8];
        int   rTok[8];
        bool  rv[8];
        #pragma unroll
        for (int r = 0; r < 8; ++r) {
            int idx = rBase + r;
            rv[r]   = idx < count;
            rTok[r] = rowToken(idx);
            wrow[r] = GATHER ? gates[(size_t)rTok[r] * Ec + expert] : omega[rTok[r]];
        }
        #pragma unroll
        for (int f = 0; f < 4; ++f) {
            int colg  = n0 + f * 16 + ln;            // all 8 elems of a lane share N
            float bia = bias[colg];
            #pragma unroll
            for (int r = 0; r < 8; ++r) {
                float v = wrow[r] * gelu_exact(acc[mf][f][r] + bia);
                size_t o = (size_t)rTok[r] * Dc + colg;
                if (GATHER) {
                    if (rv[r]) out[o] += v;          // row unique within this launch
                } else {
                    out[o] = v;                      // dense universal pass: plain write
                }
            }
        }
    }
}

// ---------------------------------------------------------------- launcher
extern "C" void kernel_launch(void* const* d_in, const int* in_sizes, int n_in,
                              void* d_out, int out_size, void* d_ws, size_t ws_size,
                              hipStream_t stream) {
    const float* x          = (const float*)d_in[0];
    const int*   task_ids   = (const int*)  d_in[1];
    const float* task_embed = (const float*)d_in[2];
    const float* W_experts  = (const float*)d_in[3];
    const float* b_experts  = (const float*)d_in[4];
    const float* W_univ     = (const float*)d_in[5];
    const float* b_univ     = (const float*)d_in[6];
    const float* W_gate     = (const float*)d_in[7];
    const float* b_gate     = (const float*)d_in[8];

    float* out_final  = (float*)d_out;                       // [NTOK][512]
    float* out_logits = out_final + (size_t)NTOK * Dc;       // [NTOK][8]

    // workspace layout (bytes)
    unsigned char* ws = (unsigned char*)d_ws;
    const size_t off_xb    = 0;                               // bf16 x: 16 MB
    const size_t off_wt    = off_xb + (size_t)NTOK * Dc * 2;  // bf16 W^T x9: 4.5 MB
    const size_t off_gates = off_wt + (size_t)9 * Dc * Dc * 2;
    const size_t off_omega = off_gates + (size_t)NTOK * Ec * 4;
    const size_t off_cnt   = off_omega + (size_t)NTOK * 4;
    const size_t off_list  = off_cnt + 256;
    __hip_bfloat16* xb    = (__hip_bfloat16*)(ws + off_xb);
    __hip_bfloat16* wt    = (__hip_bfloat16*)(ws + off_wt);
    float*          gates = (float*)(ws + off_gates);
    float*          omg   = (float*)(ws + off_omega);
    unsigned*       cnts  = (unsigned*)(ws + off_cnt);
    int*            tlist = (int*)(ws + off_list);            // [8][NTOK]

    k_convert_x<<<(NTOK * Dc) / 1024, 256, 0, stream>>>(x, xb);
    k_convert_w<<<(9 * Dc * Dc) / 256, 256, 0, stream>>>(W_experts, W_univ, wt);
    k_gate<<<(NTOK * Ec) / 256, 256, 0, stream>>>(x, task_ids, task_embed, W_gate, b_gate, out_logits);
    k_zero<<<1, 64, 0, stream>>>(cnts);
    k_top2<<<NTOK / 256, 256, 0, stream>>>(out_logits, gates, omg, cnts, tlist);

    dim3 grid(NTOK / BM, Dc / BN);
    const unsigned char* xbc = (const unsigned char*)xb;
    const unsigned char* wtc = (const unsigned char*)wt;
    // dense universal pass: final = omega * gelu(x @ Wu + b_univ)
    k_gemm<false><<<grid, 128, 0, stream>>>(xbc, wtc + (size_t)8 * Dc * Dc * 2,
                                            gates, omg, b_univ, cnts, tlist, 0, out_final);
    // routed expert passes (stream-serialized => deterministic accumulation order)
    for (int e = 0; e < Ec; ++e) {
        k_gemm<true><<<grid, 128, 0, stream>>>(xbc, wtc + (size_t)e * Dc * Dc * 2,
                                               gates, omg, b_experts + (size_t)e * Dc,
                                               cnts, tlist, e, out_final);
    }
}
